// MPNN_78632261256134
// MI455X (gfx1250) — matrix operand order repaired and checked by reading, not compiled, not run on hardware
//
#include <hip/hip_runtime.h>
#include <hip/hip_bf16.h>

// ---------------------------------------------------------------------------
// MPNN for MI455X (gfx1250): bf16 WMMA GEMMs + async-LDS-staged message pass.
// ---------------------------------------------------------------------------

#define N_NODES  32768
#define N_EDGES  49152
#define N_GRAPHS 1024
#define DD       64

typedef __attribute__((ext_vector_type(16))) __bf16 v16bf;
typedef __attribute__((ext_vector_type(8)))  __bf16 v8bf;
typedef __attribute__((ext_vector_type(8)))  float  v8f;

__device__ __forceinline__ __bf16 f2bf(float f) {
    unsigned u = __float_as_uint(f);
    unsigned r = (u + 0x7FFFu + ((u >> 16) & 1u)) >> 16;   // round-to-nearest-even
    unsigned short s = (unsigned short)r;
    return __builtin_bit_cast(__bf16, s);
}
__device__ __forceinline__ float bf2f(__bf16 b) {
    unsigned short s = __builtin_bit_cast(unsigned short, b);
    return __uint_as_float(((unsigned)s) << 16);
}
__device__ __forceinline__ float sigmf(float x) { return 1.0f / (1.0f + expf(-x)); }

// Monotonic uint key for float atomicMax (no NaNs expected).
__device__ __forceinline__ unsigned fkey(float f) {
    unsigned u = __float_as_uint(f);
    return (u & 0x80000000u) ? ~u : (u | 0x80000000u);
}
__device__ __forceinline__ float funkey(unsigned k) {
    unsigned u = (k & 0x80000000u) ? (k & 0x7FFFFFFFu) : ~k;
    return __uint_as_float(u);
}

// ---------------------------------------------------------------------------
// Generic bf16 WMMA GEMM: C[M,N] = A[M,K] * Bw[N,K]^T (+bias) (+=) (relu)
// One wave computes one 16x16 tile via v_wmma_f32_16x16x32_bf16.
// ---------------------------------------------------------------------------
__global__ void k_gemm_wmma(const __bf16* __restrict__ A,
                            const __bf16* __restrict__ Bw,
                            const float*  __restrict__ bias,
                            float* Cf, __bf16* Cb,
                            int M, int N, int K, int relu, int accum) {
    int ntiles = N >> 4;
    int wavesPerBlock = blockDim.x >> 5;
    int wid = blockIdx.x * wavesPerBlock + ((int)threadIdx.x >> 5);
    int total = (M >> 4) * ntiles;
    if (wid >= total) return;                      // whole-wave uniform exit
    int tm = wid / ntiles, tn = wid % ntiles;
    int lane = threadIdx.x & 31;
    int half = lane >> 4, l = lane & 15;

    v8f c = {0.f,0.f,0.f,0.f,0.f,0.f,0.f,0.f};
    // A frag: lane<16 -> row m0+l, K {0..7,16..23}; lane>=16 -> K {8..15,24..31}
    const __bf16* arow = A  + (size_t)(tm * 16 + l) * K + half * 8;
    // B frag: lane<16 -> col n0+l, K {0..7,16..23}; lane>=16 -> K {8..15,24..31}
    const __bf16* brow = Bw + (size_t)(tn * 16 + l) * K + half * 8;

    for (int k0 = 0; k0 < K; k0 += 32) {
        v8bf a_lo = *(const v8bf*)(arow + k0);
        v8bf a_hi = *(const v8bf*)(arow + k0 + 16);
        v8bf b_lo = *(const v8bf*)(brow + k0);
        v8bf b_hi = *(const v8bf*)(brow + k0 + 16);
        v16bf av = __builtin_shufflevector(a_lo, a_hi, 0,1,2,3,4,5,6,7,8,9,10,11,12,13,14,15);
        v16bf bv = __builtin_shufflevector(b_lo, b_hi, 0,1,2,3,4,5,6,7,8,9,10,11,12,13,14,15);
        c = __builtin_amdgcn_wmma_f32_16x16x32_bf16(false, av, false, bv,
                                                    (short)0, c, false, false);
    }

    int n = tn * 16 + l;
    float bb = bias ? bias[n] : 0.0f;
#pragma unroll
    for (int r = 0; r < 8; r++) {
        int m = tm * 16 + r + half * 8;            // C layout: VGPR r -> row r / 8+r
        size_t idx = (size_t)m * N + n;
        float v = c[r] + bb;
        if (accum && Cf) v += Cf[idx];
        if (relu) v = fmaxf(v, 0.0f);
        if (Cf) Cf[idx] = v;
        if (Cb) Cb[idx] = f2bf(v);
    }
}

// ---------------------------------------------------------------------------
// Edge message kernel with CDNA5 async-to-LDS double buffering.
// One wave per edge. W[e] is 64x64 bf16 (8 KB) streamed through LDS in
// 4 chunks of 16 rows (2 KB) via global_load_async_to_lds_b128 + ASYNCcnt.
// Lane l accumulates rows i with (i mod 4)==(l>>3) for columns (l&7)*8..+7,
// then shfl_xor-reduces the 4 row-phases and scatters 2 columns per lane.
// ---------------------------------------------------------------------------
#define MSG_WAVES 8
__global__ void k_edge_msg(const int* __restrict__ srcI, const int* __restrict__ dstI,
                           const float* __restrict__ outf, const __bf16* __restrict__ Wb,
                           float* agg, int E) {
    // single __shared__ pool -> LDS offset 0; per-wave 2 x 2KB double buffer
    __shared__ __align__(16) __bf16 smem[MSG_WAVES * 2 * 16 * DD];

    int wave = (blockIdx.x * blockDim.x + threadIdx.x) >> 5;
    int waveInBlk = (int)(threadIdx.x >> 5);
    int lane = threadIdx.x & 31;
    if (wave >= E) return;

    int s = srcI[wave], d = dstI[wave];
    const float*  arow  = outf + (size_t)s * DD;
    const __bf16* wrow  = Wb   + (size_t)wave * (DD * DD);
    unsigned ldsWaveByte = (unsigned)waveInBlk * 4096u;      // 2 buffers x 2048B

    int phase = lane >> 3;          // row phase 0..3
    int cg    = lane & 7;           // column group: columns cg*8 .. cg*8+7

    // async prefetch of chunk c (16 rows = 2048 contiguous bytes) into buffer c&1
    auto prefetch = [&](int c) {
        unsigned ldsBase = ldsWaveByte + (unsigned)(c & 1) * 2048u + (unsigned)lane * 16u;
        const __bf16* g = wrow + c * (16 * DD) + lane * 8;
#pragma unroll
        for (int j = 0; j < 4; j++) {
            asm volatile("global_load_async_to_lds_b128 %0, %1, off"
                         :: "v"(ldsBase + (unsigned)j * 512u),
                            "v"(g + j * 256)
                         : "memory");
        }
    };

    float acc[8];
#pragma unroll
    for (int k = 0; k < 8; k++) acc[k] = 0.0f;

    prefetch(0);
#pragma unroll
    for (int c = 0; c < 4; c++) {
        if (c < 3) {
            prefetch(c + 1);
            // in-order completion: <=4 outstanding means chunk c has landed
            asm volatile("s_wait_asynccnt 0x4" ::: "memory");
        } else {
            asm volatile("s_wait_asynccnt 0x0" ::: "memory");
        }
        const __bf16* buf = smem + (size_t)waveInBlk * 2048 + (size_t)(c & 1) * 1024;
#pragma unroll
        for (int ii = 0; ii < 4; ii++) {
            int r = ii * 4 + phase;                       // row within chunk
            float a = arow[c * 16 + r];                   // broadcast activation
            v8bf w = *(const v8bf*)(buf + r * DD + cg * 8);  // ds_load_b128
#pragma unroll
            for (int k = 0; k < 8; k++) acc[k] += a * bf2f(w[k]);
        }
    }

    // reduce the 4 row-phases (lanes l, l^8, l^16, l^24 share a column group)
#pragma unroll
    for (int k = 0; k < 8; k++) {
        acc[k] += __shfl_xor(acc[k], 8);
        acc[k] += __shfl_xor(acc[k], 16);
    }
    // lane l scatters columns cg*8 + 2*phase + {0,1}; all 64 columns covered once
    int col = cg * 8 + phase * 2;
    atomicAdd(&agg[(size_t)d * DD + col],     acc[phase * 2]);
    atomicAdd(&agg[(size_t)d * DD + col + 1], acc[phase * 2 + 1]);
}

// ---------------------------------------------------------------------------
// Small helpers / elementwise kernels
// ---------------------------------------------------------------------------
__global__ void k_convert(const float* __restrict__ s, __bf16* d, int n) {
    int i = blockIdx.x * blockDim.x + threadIdx.x;
    if (i < n) d[i] = f2bf(s[i]);
}

// rootT[n*64+k] = root_W[k*64+n]  (reference uses out @ root_W, not W^T)
__global__ void k_convT64(const float* __restrict__ s, __bf16* d) {
    int i = blockIdx.x * blockDim.x + threadIdx.x;
    if (i < 64 * 64) { int n = i >> 6, k = i & 63; d[i] = f2bf(s[k * 64 + n]); }
}

__global__ void k_lin0(const float* __restrict__ x, const float* __restrict__ W,
                       const float* __restrict__ b, float* outf, __bf16* outb) {
    int i = blockIdx.x * blockDim.x + threadIdx.x;
    if (i >= N_NODES * DD) return;
    int n = i >> 6, d = i & 63;
    float s = b[d];
    const float* xr = x + (size_t)n * 14;
    const float* wr = W + (size_t)d * 14;
#pragma unroll
    for (int k = 0; k < 14; k++) s += xr[k] * wr[k];
    s = fmaxf(s, 0.0f);
    outf[i] = s; outb[i] = f2bf(s);
}

__global__ void k_edge1(const float* __restrict__ ea, const float* __restrict__ W,
                        const float* __restrict__ b, __bf16* h1) {
    int i = blockIdx.x * blockDim.x + threadIdx.x;
    if (i >= N_EDGES * 128) return;
    int e = i >> 7, j = i & 127;
    const float* er = ea + (size_t)e * 4;
    const float* wr = W + (size_t)j * 4;
    float s = b[j] + er[0]*wr[0] + er[1]*wr[1] + er[2]*wr[2] + er[3]*wr[3];
    h1[i] = f2bf(fmaxf(s, 0.0f));
}

__global__ void k_count(const int* __restrict__ dstI, float* cnt, int E) {
    int e = blockIdx.x * blockDim.x + threadIdx.x;
    if (e < E) atomicAdd(&cnt[dstI[e]], 1.0f);
}

// m = relu(agg/cnt + pre_root + conv_b) -> bf16
__global__ void k_node_update(const float* __restrict__ agg, const float* __restrict__ cnt,
                              const float* __restrict__ pre, const float* __restrict__ cb,
                              __bf16* mb) {
    int i = blockIdx.x * blockDim.x + threadIdx.x;
    if (i >= N_NODES * DD) return;
    int n = i >> 6, d = i & 63;
    float c = fmaxf(cnt[n], 1.0f);
    mb[i] = f2bf(fmaxf(agg[i] / c + pre[i] + cb[d], 0.0f));
}

// GRU cell elementwise
__global__ void k_gru(const float* __restrict__ gi, const float* __restrict__ gh,
                      float* h, __bf16* hb) {
    int i = blockIdx.x * blockDim.x + threadIdx.x;
    if (i >= N_NODES * DD) return;
    int n = i >> 6, d = i & 63;
    size_t base = (size_t)n * (3 * DD) + d;
    float r  = sigmf(gi[base]          + gh[base]);
    float z  = sigmf(gi[base + DD]     + gh[base + DD]);
    float nn = tanhf(gi[base + 2 * DD] + r * gh[base + 2 * DD]);
    float hv = (1.0f - z) * nn + z * h[i];
    h[i] = hv; hb[i] = f2bf(hv);
}

// LSTM cell elementwise (Set2Set)
__global__ void k_lstm(const float* __restrict__ g, float* hl, float* cl, __bf16* hlb) {
    int i = blockIdx.x * blockDim.x + threadIdx.x;
    if (i >= N_GRAPHS * DD) return;
    int b = i >> 6, d = i & 63;
    size_t base = (size_t)b * (4 * DD) + d;
    float i_ = g[base], f_ = g[base + DD], g_ = g[base + 2 * DD], o_ = g[base + 3 * DD];
    float c = sigmf(f_) * cl[i] + sigmf(i_) * tanhf(g_);
    float h = sigmf(o_) * tanhf(c);
    cl[i] = c; hl[i] = h; hlb[i] = f2bf(h);
}

__global__ void k_attn_dot(const float* __restrict__ outf, const float* __restrict__ hl,
                           const int* __restrict__ batch, float* e_arr, unsigned* mxk) {
    int n = blockIdx.x * blockDim.x + threadIdx.x;
    if (n >= N_NODES) return;
    int b = batch[n];
    const float* xr = outf + (size_t)n * DD;
    const float* qr = hl   + (size_t)b * DD;
    float s = 0.f;
#pragma unroll
    for (int k = 0; k < DD; k++) s += xr[k] * qr[k];
    e_arr[n] = s;
    atomicMax(&mxk[b], fkey(s));
}

__global__ void k_attn_exp(const float* __restrict__ e_arr, const unsigned* __restrict__ mxk,
                           const int* __restrict__ batch, float* ex, float* den) {
    int n = blockIdx.x * blockDim.x + threadIdx.x;
    if (n >= N_NODES) return;
    int b = batch[n];
    float v = expf(e_arr[n] - funkey(mxk[b]));
    ex[n] = v;
    atomicAdd(&den[b], v);
}

__global__ void k_attn_acc(const float* __restrict__ ex, const float* __restrict__ den,
                           const int* __restrict__ batch, const float* __restrict__ outf,
                           float* r_read) {
    int i = blockIdx.x * blockDim.x + threadIdx.x;
    if (i >= N_NODES * DD) return;
    int n = i >> 6, d = i & 63;
    int b = batch[n];
    float a = ex[n] / den[b];
    atomicAdd(&r_read[(size_t)b * DD + d], a * outf[i]);
}

__global__ void k_qstar(const float* __restrict__ hl, const float* __restrict__ r_read,
                        float* qs, __bf16* qsb) {
    int i = blockIdx.x * blockDim.x + threadIdx.x;
    if (i >= N_GRAPHS * 2 * DD) return;
    int b = i >> 7, j = i & 127;
    float v = (j < DD) ? hl[(size_t)b * DD + j] : r_read[(size_t)b * DD + j - DD];
    qs[i] = v; qsb[i] = f2bf(v);
}

__global__ void k_fc2(const float* __restrict__ hid, const float* __restrict__ W,
                      const float* __restrict__ b, float* out) {
    int g = blockIdx.x * blockDim.x + threadIdx.x;
    if (g >= N_GRAPHS) return;
    const float* hr = hid + (size_t)g * 128;
    float s = b[0];
#pragma unroll
    for (int k = 0; k < 128; k++) s += hr[k] * W[k];
    out[g] = s;
}

// ---------------------------------------------------------------------------
// Host orchestration
// ---------------------------------------------------------------------------
static inline void launch_gemm(const __bf16* A, const __bf16* B, const float* bias,
                               float* Cf, __bf16* Cb, int M, int N, int K,
                               int relu, int accum, hipStream_t s) {
    int total = (M / 16) * (N / 16);
    int blocks = (total + 7) / 8;                  // 8 waves / 256-thread block
    k_gemm_wmma<<<blocks, 256, 0, s>>>(A, B, bias, Cf, Cb, M, N, K, relu, accum);
}

extern "C" void kernel_launch(void* const* d_in, const int* in_sizes, int n_in,
                              void* d_out, int out_size, void* d_ws, size_t ws_size,
                              hipStream_t stream) {
    (void)in_sizes; (void)n_in; (void)out_size; (void)ws_size;
    const float* x        = (const float*)d_in[0];
    const int*   eidx     = (const int*)  d_in[1];
    const float* eattr    = (const float*)d_in[2];
    const int*   batch    = (const int*)  d_in[3];
    const float* lin0_W   = (const float*)d_in[4];
    const float* lin0_b   = (const float*)d_in[5];
    const float* e1_W     = (const float*)d_in[6];
    const float* e1_b     = (const float*)d_in[7];
    const float* e2_W     = (const float*)d_in[8];
    const float* e2_b     = (const float*)d_in[9];
    const float* root_W   = (const float*)d_in[10];
    const float* conv_b   = (const float*)d_in[11];
    const float* gru_Wih  = (const float*)d_in[12];
    const float* gru_Whh  = (const float*)d_in[13];
    const float* gru_bih  = (const float*)d_in[14];
    const float* gru_bhh  = (const float*)d_in[15];
    const float* lstm_Wih = (const float*)d_in[16];
    const float* lstm_Whh = (const float*)d_in[17];
    const float* lstm_bih = (const float*)d_in[18];
    const float* lstm_bhh = (const float*)d_in[19];
    const float* fc1_W    = (const float*)d_in[20];
    const float* fc1_b    = (const float*)d_in[21];
    const float* fc2_W    = (const float*)d_in[22];
    const float* fc2_b    = (const float*)d_in[23];
    float* out_final = (float*)d_out;

    const int* srcI = eidx;
    const int* dstI = eidx + N_EDGES;

    // ---- workspace bump allocator ----
    unsigned char* wsp = (unsigned char*)d_ws;
    size_t off = 0;
    auto alloc = [&](size_t bytes) -> void* {
        void* p = wsp + off;
        off = (off + bytes + 255) & ~(size_t)255;
        return p;
    };
    __bf16* Wb       = (__bf16*)alloc((size_t)N_EDGES * DD * DD * 2);   // 402 MB
    __bf16* h1b      = (__bf16*)alloc((size_t)N_EDGES * 128 * 2);
    float*  outf     = (float*) alloc((size_t)N_NODES * DD * 4);
    __bf16* outb     = (__bf16*)alloc((size_t)N_NODES * DD * 2);
    float*  agg      = (float*) alloc((size_t)N_NODES * DD * 4);
    float*  pre      = (float*) alloc((size_t)N_NODES * DD * 4);
    __bf16* mb       = (__bf16*)alloc((size_t)N_NODES * DD * 2);
    float*  gi       = (float*) alloc((size_t)N_NODES * 3 * DD * 4);
    float*  gh       = (float*) alloc((size_t)N_NODES * 3 * DD * 4);
    float*  cnt      = (float*) alloc((size_t)N_NODES * 4);
    __bf16* e2Wb     = (__bf16*)alloc((size_t)4096 * 128 * 2);
    __bf16* rootTb   = (__bf16*)alloc((size_t)DD * DD * 2);
    __bf16* gWihb    = (__bf16*)alloc((size_t)3 * DD * DD * 2);
    __bf16* gWhhb    = (__bf16*)alloc((size_t)3 * DD * DD * 2);
    __bf16* lWihb    = (__bf16*)alloc((size_t)4 * DD * 2 * DD * 2);
    __bf16* lWhhb    = (__bf16*)alloc((size_t)4 * DD * DD * 2);
    __bf16* fc1Wb    = (__bf16*)alloc((size_t)128 * 128 * 2);
    float*  hl       = (float*) alloc((size_t)N_GRAPHS * DD * 4);
    float*  cl       = (float*) alloc((size_t)N_GRAPHS * DD * 4);
    __bf16* hlb      = (__bf16*)alloc((size_t)N_GRAPHS * DD * 2);
    float*  qs       = (float*) alloc((size_t)N_GRAPHS * 2 * DD * 4);
    __bf16* qsb      = (__bf16*)alloc((size_t)N_GRAPHS * 2 * DD * 2);
    float*  glstm    = (float*) alloc((size_t)N_GRAPHS * 4 * DD * 4);
    float*  e_arr    = (float*) alloc((size_t)N_NODES * 4);
    float*  ex_arr   = (float*) alloc((size_t)N_NODES * 4);
    unsigned* mxk    = (unsigned*)alloc((size_t)N_GRAPHS * 4);
    float*  den      = (float*) alloc((size_t)N_GRAPHS * 4);
    float*  r_read   = (float*) alloc((size_t)N_GRAPHS * DD * 4);
    float*  hid      = (float*) alloc((size_t)N_GRAPHS * 128 * 4);

    const int T = 256;
#define GRID(n) dim3(((n) + T - 1) / T)

    // ---- weight conversions to bf16 ----
    k_convert<<<GRID(4096*128), T, 0, stream>>>(e2_W,     e2Wb, 4096*128);
    k_convert<<<GRID(3*DD*DD),  T, 0, stream>>>(gru_Wih,  gWihb, 3*DD*DD);
    k_convert<<<GRID(3*DD*DD),  T, 0, stream>>>(gru_Whh,  gWhhb, 3*DD*DD);
    k_convert<<<GRID(4*DD*2*DD),T, 0, stream>>>(lstm_Wih, lWihb, 4*DD*2*DD);
    k_convert<<<GRID(4*DD*DD),  T, 0, stream>>>(lstm_Whh, lWhhb, 4*DD*DD);
    k_convert<<<GRID(128*128),  T, 0, stream>>>(fc1_W,    fc1Wb, 128*128);
    k_convT64<<<GRID(DD*DD),    T, 0, stream>>>(root_W,   rootTb);

    // ---- lin0 + edge MLP ----
    k_lin0 <<<GRID(N_NODES*DD), T, 0, stream>>>(x, lin0_W, lin0_b, outf, outb);
    k_edge1<<<GRID(N_EDGES*128),T, 0, stream>>>(eattr, e1_W, e1_b, h1b);
    // W = (relu(eattr@e1^T+b1)) @ e2^T + e2_b  -> bf16 [E, 4096]   (WMMA)
    launch_gemm(h1b, e2Wb, e2_b, nullptr, Wb, N_EDGES, 4096, 128, 0, 0, stream);

    // ---- in-degree counts ----
    hipMemsetAsync(cnt, 0, (size_t)N_NODES * 4, stream);
    k_count<<<GRID(N_EDGES), T, 0, stream>>>(dstI, cnt, N_EDGES);

    // ---- 3 message-passing + GRU iterations ----
    for (int it = 0; it < 3; it++) {
        hipMemsetAsync(agg, 0, (size_t)N_NODES * DD * 4, stream);
        k_edge_msg<<<GRID(N_EDGES*32), T, 0, stream>>>(srcI, dstI, outf, Wb, agg, N_EDGES);
        launch_gemm(outb, rootTb, nullptr, pre, nullptr, N_NODES, DD, DD, 0, 0, stream);
        k_node_update<<<GRID(N_NODES*DD), T, 0, stream>>>(agg, cnt, pre, conv_b, mb);
        launch_gemm(mb,   gWihb, gru_bih, gi, nullptr, N_NODES, 3*DD, DD, 0, 0, stream);
        launch_gemm(outb, gWhhb, gru_bhh, gh, nullptr, N_NODES, 3*DD, DD, 0, 0, stream);
        k_gru<<<GRID(N_NODES*DD), T, 0, stream>>>(gi, gh, outf, outb);
    }

    // ---- Set2Set (3 steps) ----
    hipMemsetAsync(hl,  0, (size_t)N_GRAPHS * DD * 4, stream);
    hipMemsetAsync(cl,  0, (size_t)N_GRAPHS * DD * 4, stream);
    hipMemsetAsync(hlb, 0, (size_t)N_GRAPHS * DD * 2, stream);
    hipMemsetAsync(qs,  0, (size_t)N_GRAPHS * 2 * DD * 4, stream);
    hipMemsetAsync(qsb, 0, (size_t)N_GRAPHS * 2 * DD * 2, stream);
    for (int st = 0; st < 3; st++) {
        launch_gemm(qsb, lWihb, lstm_bih, glstm, nullptr, N_GRAPHS, 4*DD, 2*DD, 0, 0, stream);
        launch_gemm(hlb, lWhhb, lstm_bhh, glstm, nullptr, N_GRAPHS, 4*DD, DD,   0, 1, stream);
        k_lstm<<<GRID(N_GRAPHS*DD), T, 0, stream>>>(glstm, hl, cl, hlb);
        hipMemsetAsync(mxk,    0, (size_t)N_GRAPHS * 4, stream);
        hipMemsetAsync(den,    0, (size_t)N_GRAPHS * 4, stream);
        hipMemsetAsync(r_read, 0, (size_t)N_GRAPHS * DD * 4, stream);
        k_attn_dot<<<GRID(N_NODES), T, 0, stream>>>(outf, hl, batch, e_arr, mxk);
        k_attn_exp<<<GRID(N_NODES), T, 0, stream>>>(e_arr, mxk, batch, ex_arr, den);
        k_attn_acc<<<GRID(N_NODES*DD), T, 0, stream>>>(ex_arr, den, batch, outf, r_read);
        k_qstar<<<GRID(N_GRAPHS*2*DD), T, 0, stream>>>(hl, r_read, qs, qsb);
    }

    // ---- readout ----
    launch_gemm(qsb, fc1Wb, fc1_b, hid, nullptr, N_GRAPHS, 128, 2*DD, 1, 0, stream);
    k_fc2<<<GRID(N_GRAPHS), T, 0, stream>>>(hid, fc2_W, fc2_b, out_final);
#undef GRID
}
